// OPE_72688026518220
// MI455X (gfx1250) — matrix operand order repaired and checked by reading, not compiled, not run on hardware
//
#include <hip/hip_runtime.h>
#include <hip/hip_bf16.h>

// ---------------------------------------------------------------------------
// Types for CDNA5 WMMA (wave32)
// ---------------------------------------------------------------------------
typedef __bf16 bf16_t;
typedef __attribute__((ext_vector_type(16))) __bf16 v16bf;
typedef __attribute__((ext_vector_type(8)))  __bf16 v8bf;
typedef __attribute__((ext_vector_type(8)))  float  v8f;

#define BM 128
#define BN 128
#define BK 32

// CDNA5 async direct-to-LDS path (ASYNCcnt) if the toolchain exposes it.
#if defined(__has_builtin)
# if __has_builtin(__builtin_amdgcn_global_load_async_to_lds_b128)
#  define ASYNC_LDS 1
# endif
# if __has_builtin(__builtin_amdgcn_s_wait_asynccnt)
#  define HAVE_WAIT_ASYNC_BUILTIN 1
# endif
#endif

#if defined(ASYNC_LDS)
typedef int v4i_raw __attribute__((vector_size(16)));
typedef __attribute__((address_space(1))) v4i_raw* gas_t;
typedef __attribute__((address_space(3))) v4i_raw* las_t;
#endif

__device__ __forceinline__ void wait_async0() {
#if defined(ASYNC_LDS)
# if defined(HAVE_WAIT_ASYNC_BUILTIN)
    __builtin_amdgcn_s_wait_asynccnt(0);
# else
    asm volatile("s_wait_asynccnt 0x0" ::: "memory");
# endif
#endif
}

// ---------------------------------------------------------------------------
// fp32 -> bf16 convert (activations)
// ---------------------------------------------------------------------------
__global__ void cvt_f32_bf16(const float* __restrict__ in, bf16_t* __restrict__ out, long n) {
    long i = (long)blockIdx.x * blockDim.x + threadIdx.x;
    if (i < n) out[i] = (bf16_t)in[i];
}

// fp32 W [E][K][N]  ->  bf16 Wt [E][N][K]  (transposed for contiguous-K tiles)
__global__ void transpose_to_bf16(const float* __restrict__ in, bf16_t* __restrict__ out,
                                  int E, int K, int N) {
    long i = (long)blockIdx.x * blockDim.x + threadIdx.x;
    long kn = (long)K * N;
    if (i >= (long)E * kn) return;
    int e = (int)(i / kn);
    long r = i - (long)e * kn;
    int k = (int)(r / N);
    int n = (int)(r - (long)k * N);
    out[(long)e * kn + (long)n * K + k] = (bf16_t)in[i];
}

// ---------------------------------------------------------------------------
// Batched-expert GEMM:  C[e] = act( A[e] (MxK, bf16) * Wt[e]^T + bias[e] )
// Wt is stored [E][N][K] bf16.  256 threads = 8 waves; block tile 128x128;
// wave tile 64x32 = 4x2 WMMA 16x16x32 bf16 fragments, f32 accumulate.
// Double-buffered LDS; tile staging via GLOBAL_LOAD_ASYNC_TO_LDS_B128 when
// available, else register-staged global_load_b128 + ds_store_b128.
// ---------------------------------------------------------------------------
__global__ __launch_bounds__(256)
void gemm_bf16_wmma(const bf16_t* __restrict__ A, long sAe,
                    const bf16_t* __restrict__ Wt,
                    const float*  __restrict__ bias,
                    bf16_t* __restrict__ C, long sCe,
                    int M, int N, int K, int relu)
{
    __shared__ __align__(16) bf16_t As[2][BM * BK];
    __shared__ __align__(16) bf16_t Bs[2][BN * BK];

    const int e = blockIdx.z;
    A    += (long)e * sAe;
    Wt   += (long)e * (long)N * K;
    bias += (long)e * N;
    C    += (long)e * sCe;

    const int mBase = blockIdx.x * BM;
    const int nBase = blockIdx.y * BN;
    const int tid   = threadIdx.x;
    const int lane  = tid & 31;
    const int wave  = tid >> 5;
    const int wm    = (wave & 1) * 64;   // 2 waves tile M
    const int wn    = (wave >> 1) * 32;  // 4 waves tile N
    const int l16   = lane & 15;
    const int aKlo  = (lane < 16) ? 0 : 8;   // A frag: lo lanes K0..7/16..23, hi lanes K8..15/24..31
    const int bKlo  = (lane < 16) ? 0 : 8;   // B frag: lo lanes K0..7/16..23, hi lanes K8..15/24..31

    // staging coords: 512 x 16B chunks per tile, 2 per thread (rows r0, r0+64)
    const int r0  = tid >> 2;
    const int cb0 = (tid & 3) * 8;

    v8f acc[4][2];
#pragma unroll
    for (int mt = 0; mt < 4; ++mt)
#pragma unroll
        for (int nt = 0; nt < 2; ++nt) acc[mt][nt] = {};

#if defined(ASYNC_LDS)
    auto stage_async = [&](int k0, int buf) {
        // A tile: always in-bounds (M % 128 == 0, K % 32 == 0)
        __builtin_amdgcn_global_load_async_to_lds_b128(
            (gas_t)(A + (long)(mBase + r0) * K + k0 + cb0),
            (las_t)(&As[buf][r0 * BK + cb0]), 0, 0);
        __builtin_amdgcn_global_load_async_to_lds_b128(
            (gas_t)(A + (long)(mBase + r0 + 64) * K + k0 + cb0),
            (las_t)(&As[buf][(r0 + 64) * BK + cb0]), 0, 0);
        // W tile rows, zero-pad n >= N
#pragma unroll
        for (int i = 0; i < 2; ++i) {
            int row = r0 + i * 64;
            if (nBase + row < N) {
                __builtin_amdgcn_global_load_async_to_lds_b128(
                    (gas_t)(Wt + (long)(nBase + row) * K + k0 + cb0),
                    (las_t)(&Bs[buf][row * BK + cb0]), 0, 0);
            } else {
                v8bf z;
#pragma unroll
                for (int j = 0; j < 8; ++j) z[j] = (bf16_t)0.0f;
                *(v8bf*)(&Bs[buf][row * BK + cb0]) = z;
            }
        }
    };
#else
    v8bf ra0, ra1, rb0, rb1;
    auto load_regs = [&](int k0) {
        ra0 = *(const v8bf*)(A + (long)(mBase + r0) * K + k0 + cb0);
        ra1 = *(const v8bf*)(A + (long)(mBase + r0 + 64) * K + k0 + cb0);
        if (nBase + r0 < N)
            rb0 = *(const v8bf*)(Wt + (long)(nBase + r0) * K + k0 + cb0);
        else
#pragma unroll
            for (int j = 0; j < 8; ++j) rb0[j] = (bf16_t)0.0f;
        if (nBase + r0 + 64 < N)
            rb1 = *(const v8bf*)(Wt + (long)(nBase + r0 + 64) * K + k0 + cb0);
        else
#pragma unroll
            for (int j = 0; j < 8; ++j) rb1[j] = (bf16_t)0.0f;
        __builtin_prefetch(A + (long)(mBase + r0) * K + k0 + BK, 0, 0);
    };
    auto store_regs = [&](int buf) {
        *(v8bf*)(&As[buf][r0 * BK + cb0])        = ra0;
        *(v8bf*)(&As[buf][(r0 + 64) * BK + cb0]) = ra1;
        *(v8bf*)(&Bs[buf][r0 * BK + cb0])        = rb0;
        *(v8bf*)(&Bs[buf][(r0 + 64) * BK + cb0]) = rb1;
    };
#endif

    const int nIter = K / BK;

    // prologue: stage tile 0 into buffer 0
#if defined(ASYNC_LDS)
    stage_async(0, 0);
#else
    load_regs(0);
    store_regs(0);
#endif

    for (int it = 0; it < nIter; ++it) {
        const int  k0   = it * BK;
        const int  buf  = it & 1;
        const bool next = (it + 1 < nIter);

#if defined(ASYNC_LDS)
        wait_async0();          // our async writes (into this buf) are in LDS
        __syncthreads();        // everyone's writes visible; prior reads done
        if (next) stage_async(k0 + BK, buf ^ 1);
#else
        __syncthreads();        // prior ds_stores into this buf visible
        if (next) load_regs(k0 + BK);
#endif

        v16bf afrag[4], bfrag[2];
#pragma unroll
        for (int mt = 0; mt < 4; ++mt) {
            int r = wm + mt * 16 + l16;
            v8bf lo = *(const v8bf*)(&As[buf][r * BK + aKlo]);
            v8bf hi = *(const v8bf*)(&As[buf][r * BK + aKlo + 16]);
            afrag[mt] = __builtin_shufflevector(lo, hi, 0,1,2,3,4,5,6,7,8,9,10,11,12,13,14,15);
        }
#pragma unroll
        for (int nt = 0; nt < 2; ++nt) {
            int c = wn + nt * 16 + l16;
            v8bf lo = *(const v8bf*)(&Bs[buf][c * BK + bKlo]);
            v8bf hi = *(const v8bf*)(&Bs[buf][c * BK + bKlo + 16]);
            bfrag[nt] = __builtin_shufflevector(lo, hi, 0,1,2,3,4,5,6,7,8,9,10,11,12,13,14,15);
        }
#pragma unroll
        for (int mt = 0; mt < 4; ++mt)
#pragma unroll
            for (int nt = 0; nt < 2; ++nt)
                acc[mt][nt] = __builtin_amdgcn_wmma_f32_16x16x32_bf16(
                    false, afrag[mt], false, bfrag[nt],
                    (short)0, acc[mt][nt], false, false);

#if !defined(ASYNC_LDS)
        if (next) store_regs(buf ^ 1);
#endif
    }

    // ---- epilogue: bias + optional ReLU, store bf16
    const int rowAdd = (lane < 16) ? 0 : 8;
#pragma unroll
    for (int mt = 0; mt < 4; ++mt) {
#pragma unroll
        for (int nt = 0; nt < 2; ++nt) {
            int col = nBase + wn + nt * 16 + l16;
            if (col < N) {
                float bv = bias[col];
#pragma unroll
                for (int r = 0; r < 8; ++r) {
                    int row = mBase + wm + mt * 16 + rowAdd + r;
                    float v = acc[mt][nt][r] + bv;
                    if (relu) v = fmaxf(v, 0.0f);
                    C[(long)row * N + col] = (bf16_t)v;
                }
            }
        }
    }
}

// ---------------------------------------------------------------------------
// Per-row gate: softmax(x @ W + b).  X bf16 [B][K], W fp32 [K][O], O <= 12.
// ---------------------------------------------------------------------------
__global__ void gate_softmax(const bf16_t* __restrict__ X, const float* __restrict__ W,
                             const float* __restrict__ bvec, float* __restrict__ G,
                             int Bn, int K, int O)
{
    int b = blockIdx.x * blockDim.x + threadIdx.x;
    if (b >= Bn) return;
    float acc[12];
    for (int o = 0; o < O; ++o) acc[o] = bvec[o];
    const bf16_t* xr = X + (long)b * K;
    for (int k = 0; k < K; ++k) {
        float xv = (float)xr[k];
        const float* wr = W + (long)k * O;
        for (int o = 0; o < O; ++o) acc[o] += xv * wr[o];
    }
    float m = acc[0];
    for (int o = 1; o < O; ++o) m = fmaxf(m, acc[o]);
    float s = 0.f;
    for (int o = 0; o < O; ++o) { acc[o] = expf(acc[o] - m); s += acc[o]; }
    float inv = 1.f / s;
    for (int o = 0; o < O; ++o) G[(long)b * O + o] = acc[o] * inv;
}

// ---------------------------------------------------------------------------
// Gated mixture: Out[b][c] = sum_e g[b][e] * EO[e][b][c] over up to 3 groups.
// ---------------------------------------------------------------------------
__global__ void mix_experts(const bf16_t* __restrict__ e0, const bf16_t* __restrict__ e1,
                            const bf16_t* __restrict__ e2, int n0, int n1, int n2,
                            const float* __restrict__ G, bf16_t* __restrict__ Out,
                            int Bn, int Ocols)
{
    long idx = (long)blockIdx.x * blockDim.x + threadIdx.x;
    if (idx >= (long)Bn * Ocols) return;
    int b = (int)(idx / Ocols);
    int c = (int)(idx - (long)b * Ocols);
    int O = n0 + n1 + n2;
    const float* g = G + (long)b * O;
    float acc = 0.f;
    int gi = 0;
    for (int e = 0; e < n0; ++e) acc += g[gi++] * (float)e0[((long)e * Bn + b) * Ocols + c];
    for (int e = 0; e < n1; ++e) acc += g[gi++] * (float)e1[((long)e * Bn + b) * Ocols + c];
    for (int e = 0; e < n2; ++e) acc += g[gi++] * (float)e2[((long)e * Bn + b) * Ocols + c];
    Out[idx] = (bf16_t)acc;
}

// ---------------------------------------------------------------------------
// Tower head: y = sigmoid(h @ w2 + b2), h bf16 [B][Hd], w2 fp32 [Hd]
// ---------------------------------------------------------------------------
__global__ void tower_head(const bf16_t* __restrict__ H, const float* __restrict__ w,
                           const float* __restrict__ b, float* __restrict__ Y, int Bn, int Hd)
{
    int i = blockIdx.x * blockDim.x + threadIdx.x;
    if (i >= Bn) return;
    float acc = b[0];
    const bf16_t* hr = H + (long)i * Hd;
    for (int j = 0; j < Hd; ++j) acc += (float)hr[j] * w[j];
    Y[i] = 1.f / (1.f + expf(-acc));
}

// ---------------------------------------------------------------------------
// Host orchestration
// ---------------------------------------------------------------------------
extern "C" void kernel_launch(void* const* d_in, const int* in_sizes, int n_in,
                              void* d_out, int out_size, void* d_ws, size_t ws_size,
                              hipStream_t stream)
{
    (void)out_size; (void)ws_size;

    // Detect flatten order: x (B*512 elems, uniquely huge) first, or params tree first.
    int xi = (in_sizes[0] >= (1 << 22)) ? 0 : (n_in - 3);
    int pbase = (xi == 0) ? 3 : 0;
    const int Bn = in_sizes[xi] / 512;

    auto P = [&](int k) -> const float* { return (const float*)d_in[pbase + k]; };
    const float* x  = (const float*)d_in[xi];
    const float* x1 = (const float*)d_in[xi + 1];
    const float* x2 = (const float*)d_in[xi + 2];

    // Param tree indices (sorted-key flatten):
    // 0 emb_full.b 1 emb_full.w 2 emb_t1.b 3 emb_t1.w 4 emb_t2.b 5 emb_t2.w
    // per-layer base 6+18*l: 0 gs.b 1 gs.w 2 g1.b 3 g1.w 4 g2.b 5 g2.w
    //   6 s.b1 7 s.b2 8 s.w1 9 s.w2  10 t1.b1 11 t1.b2 12 t1.w1 13 t1.w2
    //   14 t2.b1 15 t2.b2 16 t2.w1 17 t2.w2
    // 60..63 tower1 fc1.b fc1.w fc2.b fc2.w ; 64..67 tower2

    char*  base = (char*)d_ws;
    size_t off  = 0;
    auto alloc = [&](size_t bytes) -> void* {
        off = (off + 255) & ~(size_t)255;
        void* p = base + off;
        off += bytes;
        return p;
    };

    bf16_t* wEf = (bf16_t*)alloc((size_t)256 * 512 * 2);
    bf16_t* wE1 = (bf16_t*)alloc((size_t)128 * 64 * 2);
    bf16_t* wE2 = (bf16_t*)alloc((size_t)128 * 64 * 2);
    bf16_t *wS1[3], *wS2[3], *wT11[3], *wT12[3], *wT21[3], *wT22[3];
    for (int l = 0; l < 3; ++l) {
        int inF = l ? 128 : 256;
        wS1[l]  = (bf16_t*)alloc((size_t)4 * 256 * inF * 2);
        wS2[l]  = (bf16_t*)alloc((size_t)4 * 128 * 256 * 2);
        wT11[l] = (bf16_t*)alloc((size_t)4 * 256 * 128 * 2);
        wT12[l] = (bf16_t*)alloc((size_t)4 * 128 * 256 * 2);
        wT21[l] = (bf16_t*)alloc((size_t)4 * 256 * 128 * 2);
        wT22[l] = (bf16_t*)alloc((size_t)4 * 128 * 256 * 2);
    }
    bf16_t* wTw1 = (bf16_t*)alloc((size_t)64 * 128 * 2);
    bf16_t* wTw2 = (bf16_t*)alloc((size_t)64 * 128 * 2);

    bf16_t* xbf  = (bf16_t*)alloc((size_t)Bn * 512 * 2);
    bf16_t* x1bf = (bf16_t*)alloc((size_t)Bn * 64 * 2);
    bf16_t* x2bf = (bf16_t*)alloc((size_t)Bn * 64 * 2);
    bf16_t* actF[2] = { (bf16_t*)alloc((size_t)Bn * 256 * 2), (bf16_t*)alloc((size_t)Bn * 256 * 2) };
    bf16_t* act1[2] = { (bf16_t*)alloc((size_t)Bn * 128 * 2), (bf16_t*)alloc((size_t)Bn * 128 * 2) };
    bf16_t* act2[2] = { (bf16_t*)alloc((size_t)Bn * 128 * 2), (bf16_t*)alloc((size_t)Bn * 128 * 2) };
    bf16_t* hbuf = (bf16_t*)alloc((size_t)4 * Bn * 256 * 2);
    bf16_t* eoS  = (bf16_t*)alloc((size_t)4 * Bn * 128 * 2);
    bf16_t* eo1  = (bf16_t*)alloc((size_t)4 * Bn * 128 * 2);
    bf16_t* eo2  = (bf16_t*)alloc((size_t)4 * Bn * 128 * 2);
    float*  g1   = (float*)alloc((size_t)Bn * 8 * 4);
    float*  g2   = (float*)alloc((size_t)Bn * 8 * 4);
    float*  gs   = (float*)alloc((size_t)Bn * 12 * 4);
    bf16_t* th1  = (bf16_t*)alloc((size_t)Bn * 64 * 2);
    bf16_t* th2  = (bf16_t*)alloc((size_t)Bn * 64 * 2);

    auto cvtT = [&](const float* w, bf16_t* out, int E, int K, int N) {
        long n = (long)E * K * N;
        transpose_to_bf16<<<dim3((unsigned)((n + 255) / 256)), dim3(256), 0, stream>>>(w, out, E, K, N);
    };
    auto cvt = [&](const float* in, bf16_t* out, long n) {
        cvt_f32_bf16<<<dim3((unsigned)((n + 255) / 256)), dim3(256), 0, stream>>>(in, out, n);
    };
    auto gemm = [&](const bf16_t* A, long sAe, const bf16_t* Wt, const float* bias,
                    bf16_t* C, long sCe, int M, int N, int K, int E, int relu) {
        dim3 g(M / BM, (N + BN - 1) / BN, E);
        gemm_bf16_wmma<<<g, dim3(256), 0, stream>>>(A, sAe, Wt, bias, C, sCe, M, N, K, relu);
    };

    // ---- prep weights (bf16, [N][K]) and activations (bf16)
    cvtT(P(1), wEf, 1, 512, 256);
    cvtT(P(3), wE1, 1, 64, 128);
    cvtT(P(5), wE2, 1, 64, 128);
    for (int l = 0; l < 3; ++l) {
        int pb = 6 + 18 * l, inF = l ? 128 : 256;
        cvtT(P(pb + 8),  wS1[l],  4, inF, 256);
        cvtT(P(pb + 9),  wS2[l],  4, 256, 128);
        cvtT(P(pb + 12), wT11[l], 4, 128, 256);
        cvtT(P(pb + 13), wT12[l], 4, 256, 128);
        cvtT(P(pb + 16), wT21[l], 4, 128, 256);
        cvtT(P(pb + 17), wT22[l], 4, 256, 128);
    }
    cvtT(P(61), wTw1, 1, 128, 64);
    cvtT(P(65), wTw2, 1, 128, 64);

    cvt(x,  xbf,  (long)Bn * 512);
    cvt(x1, x1bf, (long)Bn * 64);
    cvt(x2, x2bf, (long)Bn * 64);

    // ---- embeddings (linear, no relu)
    gemm(xbf,  0, wEf, P(0), actF[0], 0, Bn, 256, 512, 1, 0);
    gemm(x1bf, 0, wE1, P(2), act1[0], 0, Bn, 128, 64, 1, 0);
    gemm(x2bf, 0, wE2, P(4), act2[0], 0, Bn, 128, 64, 1, 0);

    // ---- CGC layers
    int cur = 0;
    for (int l = 0; l < 3; ++l) {
        int pb = 6 + 18 * l, inF = l ? 128 : 256;
        bf16_t* xf  = actF[cur];
        bf16_t* x1a = act1[cur];
        bf16_t* x2a = act2[cur];
        long sH = (long)Bn * 256, sO = (long)Bn * 128;

        gemm(xf,   0,  wS1[l],  P(pb + 6),  hbuf, sH, Bn, 256, inF, 4, 1);
        gemm(hbuf, sH, wS2[l],  P(pb + 7),  eoS,  sO, Bn, 128, 256, 4, 1);
        gemm(x1a,  0,  wT11[l], P(pb + 10), hbuf, sH, Bn, 256, 128, 4, 1);
        gemm(hbuf, sH, wT12[l], P(pb + 11), eo1,  sO, Bn, 128, 256, 4, 1);
        gemm(x2a,  0,  wT21[l], P(pb + 14), hbuf, sH, Bn, 256, 128, 4, 1);
        gemm(hbuf, sH, wT22[l], P(pb + 15), eo2,  sO, Bn, 128, 256, 4, 1);

        gate_softmax<<<dim3((Bn + 127) / 128), dim3(128), 0, stream>>>(x1a, P(pb + 3), P(pb + 2), g1, Bn, 128, 8);
        gate_softmax<<<dim3((Bn + 127) / 128), dim3(128), 0, stream>>>(x2a, P(pb + 5), P(pb + 4), g2, Bn, 128, 8);
        if (l < 2)
            gate_softmax<<<dim3((Bn + 127) / 128), dim3(128), 0, stream>>>(xf, P(pb + 1), P(pb + 0), gs, Bn, inF, 12);

        int nxt = cur ^ 1;
        long nmix = (long)Bn * 128;
        unsigned mb = (unsigned)((nmix + 255) / 256);
        mix_experts<<<dim3(mb), dim3(256), 0, stream>>>(eo1, eoS, nullptr, 4, 4, 0, g1, act1[nxt], Bn, 128);
        mix_experts<<<dim3(mb), dim3(256), 0, stream>>>(eo2, eoS, nullptr, 4, 4, 0, g2, act2[nxt], Bn, 128);
        if (l < 2)
            mix_experts<<<dim3(mb), dim3(256), 0, stream>>>(eo1, eo2, eoS, 4, 4, 4, gs, actF[nxt], Bn, 128);
        cur = nxt;
    }

    // ---- towers
    gemm(act1[cur], 0, wTw1, P(60), th1, 0, Bn, 64, 128, 1, 1);
    gemm(act2[cur], 0, wTw2, P(64), th2, 0, Bn, 64, 128, 1, 1);
    tower_head<<<dim3((Bn + 255) / 256), dim3(256), 0, stream>>>(th1, P(63), P(62), (float*)d_out, Bn, 64);
    tower_head<<<dim3((Bn + 255) / 256), dim3(256), 0, stream>>>(th2, P(67), P(66), (float*)d_out + Bn, Bn, 64);
}
